// RQ_3599182594531
// MI455X (gfx1250) — matrix operand order repaired and checked by reading, not compiled, not run on hardware
//
#include <hip/hip_runtime.h>

// ---------------------------------------------------------------------------
// Residual VQ (3 codebooks) for MI455X / gfx1250.
//  - distance GEMM on v_wmma_f32_16x16x32_bf16 (wave32, 16x16 tiles)
//  - residual kept in fp32 registers in exactly the WMMA A-fragment layout
//  - codebooks pre-converted to bf16, transposed [D][K], in d_ws (prep kernel)
//  - B tile: one 32B LDS row per d, 48B padded stride; fragments read as 16 strided 16-bit LDS loads
//  - double-buffered LDS B-tile: one barrier per k-chunk, global-load latency
//    overlapped with the 8-WMMA compute block
//  - streaming input/output marked non-temporal to keep codebooks hot in L2
// ---------------------------------------------------------------------------

typedef __attribute__((ext_vector_type(16))) __bf16        v16bf;
typedef __attribute__((ext_vector_type(8)))  float         v8f;
typedef __attribute__((ext_vector_type(4)))  float         f32x4;
typedef __attribute__((ext_vector_type(4)))  unsigned int  u32x4;

constexpr int   N_ROWS = 262144;
constexpr int   DIM    = 256;
constexpr int   K0 = 1024, K1 = 512, K2 = 256;
constexpr float LOSS_SCALE = 1.25f / (262144.0f * 256.0f);  // (BETA+GAMMA)/(N*D)

// workspace layout (bytes)
constexpr size_t WT0_OFF  = 0;                                   // [256][1024] bf16
constexpr size_t WT1_OFF  = WT0_OFF + (size_t)DIM * K0 * 2;      // [256][512]  bf16
constexpr size_t WT2_OFF  = WT1_OFF + (size_t)DIM * K1 * 2;      // [256][256]  bf16
constexpr size_t WSQ0_OFF = WT2_OFF + (size_t)DIM * K2 * 2;      // 1024 f32
constexpr size_t WSQ1_OFF = WSQ0_OFF + (size_t)K0 * 4;           // 512 f32
constexpr size_t WSQ2_OFF = WSQ1_OFF + (size_t)K1 * 4;           // 256 f32

// ---------------------------------------------------------------------------
__global__ void rvq_zero(float* __restrict__ losses) {
    if (threadIdx.x < 3) losses[threadIdx.x] = 0.0f;
}

// One block per codebook row k: emit bf16 transposed W and |w_k|^2.
__global__ __launch_bounds__(256) void rvq_prep(
    const float* __restrict__ W0, const float* __restrict__ W1,
    const float* __restrict__ W2, unsigned char* __restrict__ ws)
{
    const int b = blockIdx.x;
    const int d = threadIdx.x;

    const float* W; __bf16* Wt; float* wsq; int K, k;
    if (b < K0)            { W = W0; Wt = (__bf16*)(ws + WT0_OFF); wsq = (float*)(ws + WSQ0_OFF); K = K0; k = b; }
    else if (b < K0 + K1)  { W = W1; Wt = (__bf16*)(ws + WT1_OFF); wsq = (float*)(ws + WSQ1_OFF); K = K1; k = b - K0; }
    else                   { W = W2; Wt = (__bf16*)(ws + WT2_OFF); wsq = (float*)(ws + WSQ2_OFF); K = K2; k = b - K0 - K1; }

    const float w = W[(size_t)k * DIM + d];
    Wt[(size_t)d * K + k] = (__bf16)w;

    float s = w * w;
#pragma unroll
    for (int m = 1; m < 32; m <<= 1) s += __shfl_xor(s, m, 32);

    __shared__ float red[8];
    if ((threadIdx.x & 31) == 0) red[threadIdx.x >> 5] = s;
    __syncthreads();
    if (threadIdx.x == 0) {
        float t = 0.0f;
#pragma unroll
        for (int i = 0; i < 8; ++i) t += red[i];
        wsq[k] = t;
    }
}

// ---------------------------------------------------------------------------
// Main fused kernel: 256 threads = 8 waves; each wave owns 16 rows.
// Per-lane residual storage r[c*16 + e] maps to column d = 32c + 16g + 8*half + i
// (g = e>>3, i = e&7), i.e. exactly the 16-bit A-matrix VGPR layout.
__global__ __launch_bounds__(256) void rvq_main(
    const float* __restrict__ input,
    const float* __restrict__ W0, const float* __restrict__ W1,
    const float* __restrict__ W2,
    float* __restrict__ out, float* __restrict__ losses,
    const unsigned char* __restrict__ ws)
{
    __shared__ __align__(16) unsigned char btile[2][256 * 48];  // 2 x 12KB, 48B rows
    __shared__ int idx_s[8][16];

    const int tid  = threadIdx.x;
    const int lane = tid & 31;
    const int wave = tid >> 5;
    const int half = lane >> 4;      // 0: d-offsets {0..7,16..23}; 1: {8..15,24..31}
    const int mrow = lane & 15;      // row within the wave tile; also score column
    const int row  = blockIdx.x * 128 + wave * 16 + mrow;

    // ---- load residual = input (fp32, in A-fragment ordering), non-temporal ----
    float r[128];
    const float* inrow = input + (size_t)row * DIM;
#pragma unroll
    for (int c = 0; c < 8; ++c)
#pragma unroll
        for (int g = 0; g < 2; ++g) {
            const int dbase = 32 * c + 16 * g + 8 * half;
            f32x4 q0 = __builtin_nontemporal_load((const f32x4*)(inrow + dbase));
            f32x4 q1 = __builtin_nontemporal_load((const f32x4*)(inrow + dbase + 4));
            const int bse = c * 16 + g * 8;
            r[bse + 0] = q0.x; r[bse + 1] = q0.y; r[bse + 2] = q0.z; r[bse + 3] = q0.w;
            r[bse + 4] = q1.x; r[bse + 5] = q1.y; r[bse + 6] = q1.z; r[bse + 7] = q1.w;
        }

    const float* Wstage[3] = { W0, W1, W2 };
    const int    Ks[3]     = { K0, K1, K2 };
    const __bf16* Wts[3]   = { (const __bf16*)(ws + WT0_OFF),
                               (const __bf16*)(ws + WT1_OFF),
                               (const __bf16*)(ws + WT2_OFF) };
    const float* wsqs[3]   = { (const float*)(ws + WSQ0_OFF),
                               (const float*)(ws + WSQ1_OFF),
                               (const float*)(ws + WSQ2_OFF) };

#pragma unroll
    for (int s = 0; s < 3; ++s) {
        const int     K   = Ks[s];
        const __bf16* Wt  = Wts[s];
        const float*  wsq = wsqs[s];

        // ---- convert residual to bf16 A-fragments (reused for all k-chunks) ----
        v16bf a[8];
#pragma unroll
        for (int c = 0; c < 8; ++c)
#pragma unroll
            for (int e = 0; e < 16; ++e)
                a[c][e] = (__bf16)r[c * 16 + e];

        float minv[8]; int mini[8];
#pragma unroll
        for (int j = 0; j < 8; ++j) { minv[j] = 3.0e38f; mini[j] = 0; }

        // ---- prologue: stage k-chunk 0 into buf0 (thread == d-row, 32B/row) ----
        const __bf16* srcbase = Wt + (size_t)tid * K;
        {
            u32x4 p0 = *(const u32x4*)(srcbase);
            u32x4 p1 = *(const u32x4*)(srcbase + 8);
            *(u32x4*)(btile[0] + tid * 48)      = p0;
            *(u32x4*)(btile[0] + tid * 48 + 16) = p1;
        }
        int cur = 0;

        // ---- k-loop: 16 codebook columns per chunk, double-buffered B tile ----
        for (int kb = 0; kb < K; kb += 16) {
            // issue next chunk's global loads (latency hidden by compute below)
            const bool has_next = (kb + 16) < K;
            u32x4 n0, n1;
            if (has_next) {
                const __bf16* src = srcbase + kb + 16;
                n0 = *(const u32x4*)(src);
                n1 = *(const u32x4*)(src + 8);
            }
            __syncthreads();   // buf[cur] stores visible; prior reads of buf[cur^1] done

            const float wsqv = wsq[kb + mrow];

            v8f acc = {};
#pragma unroll
            for (int c = 0; c < 8; ++c) {
                // B fragment: lane holds col n=mrow; element e is d = 32c+16(e>>3)+8*half+(e&7)
                union { __bf16 h[16]; v16bf b; } bu;
                const unsigned char* p = btile[cur] + (32 * c + 8 * half) * 48 + mrow * 2;
#pragma unroll
                for (int e = 0; e < 16; ++e)
                    bu.h[e] = *(const __bf16*)(p + (16 * (e >> 3) + (e & 7)) * 48);
                acc = __builtin_amdgcn_wmma_f32_16x16x32_bf16(
                          false, a[c], false, bu.b, (short)0, acc, false, false);
            }

#pragma unroll
            for (int j = 0; j < 8; ++j) {
                const float val = wsqv - 2.0f * acc[j];   // |w|^2 - 2 r.w
                if (val < minv[j]) { minv[j] = val; mini[j] = kb + mrow; }
            }

            if (has_next) {
                *(u32x4*)(btile[cur ^ 1] + tid * 48)      = n0;
                *(u32x4*)(btile[cur ^ 1] + tid * 48 + 16) = n1;
                cur ^= 1;
            }
        }

        // ---- argmin across the 16 lanes of each half (rows j+8*half) ----
#pragma unroll
        for (int j = 0; j < 8; ++j) {
            float v = minv[j]; int i = mini[j];
#pragma unroll
            for (int m = 1; m <= 8; m <<= 1) {
                const float ov = __shfl_xor(v, m, 32);
                const int   oi = __shfl_xor(i, m, 32);
                if (ov < v || (ov == v && oi < i)) { v = ov; i = oi; }
            }
            minv[j] = v; mini[j] = i;
        }
        if (mrow == 0) {
#pragma unroll
            for (int j = 0; j < 8; ++j) idx_s[wave][8 * half + j] = mini[j];
        }
        __syncthreads();
        const int kidx = idx_s[wave][mrow];
        __syncthreads();   // also separates LDS tile reuse across stages

        // ---- gather quantized row (fp32), update residual, accumulate loss ----
        const float* qrow = Wstage[s] + (size_t)kidx * DIM;
        float lsum = 0.0f;
#pragma unroll
        for (int c = 0; c < 8; ++c)
#pragma unroll
            for (int g = 0; g < 2; ++g) {
                const int dbase = 32 * c + 16 * g + 8 * half;
                f32x4 q0 = *(const f32x4*)(qrow + dbase);
                f32x4 q1 = *(const f32x4*)(qrow + dbase + 4);
                const int bse = c * 16 + g * 8;
                r[bse + 0] -= q0.x; lsum += r[bse + 0] * r[bse + 0];
                r[bse + 1] -= q0.y; lsum += r[bse + 1] * r[bse + 1];
                r[bse + 2] -= q0.z; lsum += r[bse + 2] * r[bse + 2];
                r[bse + 3] -= q0.w; lsum += r[bse + 3] * r[bse + 3];
                r[bse + 4] -= q1.x; lsum += r[bse + 4] * r[bse + 4];
                r[bse + 5] -= q1.y; lsum += r[bse + 5] * r[bse + 5];
                r[bse + 6] -= q1.z; lsum += r[bse + 6] * r[bse + 6];
                r[bse + 7] -= q1.w; lsum += r[bse + 7] * r[bse + 7];
            }
#pragma unroll
        for (int m = 1; m < 32; m <<= 1) lsum += __shfl_xor(lsum, m, 32);
        if (lane == 0) atomicAdd(losses + s, lsum * LOSS_SCALE);
    }

    // ---- output = input - residual_final (sum of quantized stages), NT ----
    float* outrow = out + (size_t)row * DIM;
#pragma unroll
    for (int c = 0; c < 8; ++c)
#pragma unroll
        for (int g = 0; g < 2; ++g) {
            const int dbase = 32 * c + 16 * g + 8 * half;
            f32x4 i0 = __builtin_nontemporal_load((const f32x4*)(inrow + dbase));
            f32x4 i1 = __builtin_nontemporal_load((const f32x4*)(inrow + dbase + 4));
            const int bse = c * 16 + g * 8;
            f32x4 o0, o1;
            o0.x = i0.x - r[bse + 0]; o0.y = i0.y - r[bse + 1];
            o0.z = i0.z - r[bse + 2]; o0.w = i0.w - r[bse + 3];
            o1.x = i1.x - r[bse + 4]; o1.y = i1.y - r[bse + 5];
            o1.z = i1.z - r[bse + 6]; o1.w = i1.w - r[bse + 7];
            __builtin_nontemporal_store(o0, (f32x4*)(outrow + dbase));
            __builtin_nontemporal_store(o1, (f32x4*)(outrow + dbase + 4));
        }
}

// ---------------------------------------------------------------------------
extern "C" void kernel_launch(void* const* d_in, const int* in_sizes, int n_in,
                              void* d_out, int out_size, void* d_ws, size_t ws_size,
                              hipStream_t stream) {
    const float* input = (const float*)d_in[0];
    const float* W0    = (const float*)d_in[1];
    const float* W1    = (const float*)d_in[2];
    const float* W2    = (const float*)d_in[3];

    float* out    = (float*)d_out;
    float* losses = out + (size_t)N_ROWS * DIM;   // outputs concatenated flat
    unsigned char* ws = (unsigned char*)d_ws;

    rvq_zero<<<1, 32, 0, stream>>>(losses);
    rvq_prep<<<K0 + K1 + K2, 256, 0, stream>>>(W0, W1, W2, ws);
    rvq_main<<<N_ROWS / 128, 256, 0, stream>>>(input, W0, W1, W2, out, losses, ws);
}
